// EEG_StandAlone_Att_30777735643239
// MI455X (gfx1250) — matrix operand order repaired and checked by reading, not compiled, not run on hardware
//
#include <hip/hip_runtime.h>
#include <hip/hip_bf16.h>

typedef __attribute__((ext_vector_type(16))) _Float16 v16h;
typedef __attribute__((ext_vector_type(8)))  _Float16 v8h;
typedef __attribute__((ext_vector_type(8)))  float    v8f;

#define DI __device__ __forceinline__

// ---------------- workspace layout (float offsets) ----------------
static constexpr size_t O_EMB0 = 0;                     // 4*1*5
static constexpr size_t O_EMB1 = 32;                    // 4*2*5
static constexpr size_t O_EMB2 = 96;                    // 4*2*5
static constexpr size_t O_QK0  = 160;                   // 1
static constexpr size_t O_VB0  = 192;                   // 64*5
static constexpr size_t O_ST0  = 512;                   // mean[64], inv[64]
static constexpr size_t O_ST1  = 768;                   // mean[128], inv[128]
static constexpr size_t O_ST2  = 1088;                  // mean[16], inv[16]
static constexpr size_t O_W1   = 1152;                  // 64*768 halves = 24576 floats
static constexpr size_t O_W2   = 25728;                 // 128*96 halves = 6144 floats
static constexpr size_t O_S0   = 32768;                 // [2,8,3000,64] f32
static constexpr size_t O_H1P  = O_S0  + 3072000ull;    // [2,6,1504,64] f16 -> 577536 floats
static constexpr size_t O_PJ1  = O_H1P + 577536ull;     // [18048,768] f32
static constexpr size_t O_S1   = O_PJ1 + 13860864ull;   // [2,4,1500,128] f32
static constexpr size_t O_H2P  = O_S1  + 1536000ull;    // [2,2,754,128] f16 -> 193024 floats (386048 reserved)
static constexpr size_t O_PJ2  = O_H2P + 386048ull;     // [3024(pad),96] f32
static constexpr size_t O_S2   = O_PJ2 + 290304ull;     // [2,1,750,16] f32 (24000)

// ---------------- tiny prep: emb softmax, qk scalar, vbar ----------------
DI void make_emb(const float* em, const float* ea, const float* eb,
                 int co, int kh, int kw, float* out) {
  float la[4][2], lb[4][5];
  for (int m = 0; m < 4; ++m) {
    for (int k = 0; k < kh; ++k) {
      float s = 0.f;
      for (int c = 0; c < co; ++c) s += em[m * co + c] * ea[c * kh + k];
      la[m][k] = s;
    }
    for (int l = 0; l < kw; ++l) {
      float s = 0.f;
      for (int c = 0; c < co; ++c) s += em[m * co + c] * eb[c * kw + l];
      lb[m][l] = s;
    }
  }
  for (int k = 0; k < kh; ++k)
    for (int l = 0; l < kw; ++l) {
      float mx = -1e30f;
      for (int m = 0; m < 4; ++m) mx = fmaxf(mx, la[m][k] + lb[m][l]);
      float e[4], sum = 0.f;
      for (int m = 0; m < 4; ++m) { e[m] = __expf(la[m][k] + lb[m][l] - mx); sum += e[m]; }
      for (int m = 0; m < 4; ++m) out[(m * kh + k) * kw + l] = e[m] / sum;
    }
}

__global__ void prep_small(const float* q0, const float* k0, const float* v0,
                           const float* ea0, const float* eb0, const float* em0,
                           const float* ea1, const float* eb1, const float* em1,
                           const float* ea2, const float* eb2, const float* em2,
                           float* ws) {
  if (threadIdx.x != 0) return;
  make_emb(em0, ea0, eb0, 64, 1, 5, ws + O_EMB0);
  make_emb(em1, ea1, eb1, 128, 2, 5, ws + O_EMB1);
  make_emb(em2, ea2, eb2, 16, 2, 5, ws + O_EMB2);
  float qk = 0.f;
  for (int o = 0; o < 64; ++o) qk += q0[o] * k0[o];
  ws[O_QK0] = qk;
  for (int o = 0; o < 64; ++o)
    for (int l = 0; l < 5; ++l) {
      float s = 0.f;
      for (int m = 0; m < 4; ++m) s += ws[O_EMB0 + m * 5 + l] * v0[m * 64 + o];
      ws[O_VB0 + o * 5 + l] = s;
    }
}

// ---------------- pack fused f16 weight matrices [K x N] ----------------
__global__ void pack_w(const float* q1, const float* k1, const float* v1,
                       const float* q2, const float* k2, const float* v2,
                       _Float16* W1, _Float16* W2) {
  int idx = blockIdx.x * 256 + threadIdx.x;     // 61440 total
  if (idx < 64 * 768) {
    int k = idx / 768, c = idx % 768;
    float v;
    if (c < 128)      v = q1[k * 128 + c];
    else if (c < 256) v = k1[k * 128 + (c - 128)];
    else { int m = (c - 256) / 128, o = (c - 256) % 128; v = v1[(m * 64 + k) * 128 + o]; }
    W1[idx] = (_Float16)v;
  } else {
    int i2 = idx - 64 * 768;                    // < 12288
    int k = i2 / 96, c = i2 % 96;
    float v;
    if (c < 16)      v = q2[k * 16 + c];
    else if (c < 32) v = k2[k * 16 + (c - 16)];
    else { int m = (c - 32) / 16, o = (c - 32) % 16; v = v2[(m * 128 + k) * 16 + o]; }
    W2[i2] = (_Float16)v;
  }
}

// ---------------- stage 0: Cin==1 collapsed attention ----------------
__global__ void stage0_attn(const float* __restrict__ x, const float* __restrict__ ws,
                            float* __restrict__ s0) {
  int wave = threadIdx.x >> 5, lane = threadIdx.x & 31;
  int pos = blockIdx.x * 8 + wave;              // b*24000 + h*3000 + w
  if (pos >= 48000) return;
  int w = pos % 3000;
  float qk = ws[O_QK0];
  const float* vb = ws + O_VB0;
  float xc = x[pos];
  float xv[5], sc[5];
#pragma unroll
  for (int j = 0; j < 5; ++j) {
    int wj = w + j - 2;
    xv[j] = (wj >= 0 && wj < 3000) ? x[pos + j - 2] : 0.f;
    sc[j] = xc * xv[j] * qk;
  }
  float mx = sc[0];
#pragma unroll
  for (int j = 1; j < 5; ++j) mx = fmaxf(mx, sc[j]);
  float se = 0.f;
#pragma unroll
  for (int j = 0; j < 5; ++j) { sc[j] = __expf(sc[j] - mx); se += sc[j]; }
  float inv = 1.f / se;
  for (int o = lane; o < 64; o += 32) {
    float acc = 0.f;
#pragma unroll
    for (int j = 0; j < 5; ++j) acc += sc[j] * inv * xv[j] * vb[o * 5 + j];
    s0[(size_t)pos * 64 + o] = acc;
  }
}

// ---------------- BN stats: one block per channel ----------------
__global__ void bn_stats(const float* __restrict__ src, int N, int C,
                         float* __restrict__ stats) {
  int c = blockIdx.x;
  float s = 0.f, s2 = 0.f;
  for (int r = threadIdx.x; r < N; r += 256) {
    float v = src[(size_t)r * C + c];
    s += v; s2 += v * v;
  }
  __shared__ float r1[256], r2[256];
  r1[threadIdx.x] = s; r2[threadIdx.x] = s2;
  __syncthreads();
  for (int off = 128; off > 0; off >>= 1) {
    if (threadIdx.x < off) { r1[threadIdx.x] += r1[threadIdx.x + off]; r2[threadIdx.x] += r2[threadIdx.x + off]; }
    __syncthreads();
  }
  if (threadIdx.x == 0) {
    float mean = r1[0] / N;
    float var = r2[0] / N - mean * mean;
    stats[c] = mean;
    stats[C + c] = rsqrtf(var + 1e-5f);
  }
}

// ---- fused BN + ReLU + stride-2 subsample + zero-pad + f16 cast (NHWC) ----
__global__ void build_pad(const float* __restrict__ src, const float* __restrict__ stats,
                          const float* __restrict__ g, const float* __restrict__ bb,
                          _Float16* __restrict__ dst,
                          int B, int Hs, int Ws, int C,
                          int Hp, int Wp, int pt, int pl, int Hv, int Wv) {
  size_t idx = (size_t)blockIdx.x * 256 + threadIdx.x;
  size_t total = (size_t)B * Hp * Wp * C;
  if (idx >= total) return;
  int o = (int)(idx % C);
  size_t t = idx / C;
  int wp = (int)(t % Wp); t /= Wp;
  int hp = (int)(t % Hp);
  int b = (int)(t / Hp);
  int h = hp - pt, w = wp - pl;
  float val = 0.f;
  if (h >= 0 && h < Hv && w >= 0 && w < Wv) {
    float v = src[(((size_t)b * Hs + 2 * h) * Ws + 2 * w) * C + o];
    v = (v - stats[o]) * stats[C + o] * g[o] + bb[o];
    val = fmaxf(v, 0.f);
  }
  dst[idx] = (_Float16)val;
}

// ---------------- WMMA f16 GEMM: C[M,N] = A[M,K] * B[K,N] ----------------
// 8 waves/block, each wave owns one 16x16 tile of a 16x128 block tile.
// A tile and transposed-B tile staged in LDS; per-lane fragments are
// contiguous 16B runs -> ds_load_b128; K templated so the loop unrolls.
template <int K>
__global__ void gemm_f16(const _Float16* __restrict__ A, const _Float16* __restrict__ Bw,
                         float* __restrict__ Cc, int Mrows, int N) {
  __shared__ alignas(16) _Float16 lA[16 * K];
  __shared__ alignas(16) _Float16 lBT[128 * K];
  constexpr int Kd = K / 2;
  int m0 = blockIdx.x * 16;
  int n0base = blockIdx.y * 128;
  // stage A tile (row-major), zero-fill past Mrows
  for (int d = threadIdx.x; d < 16 * Kd; d += 256) {
    int r = d / Kd, kd = d % Kd;
    int row = m0 + r;
    ((unsigned int*)lA)[d] =
        (row < Mrows) ? ((const unsigned int*)A)[(size_t)row * Kd + kd] : 0u;
  }
  // stage B tile transposed: lBT[c][k] = B[k][n0base + c]
  for (int e = threadIdx.x; e < 128 * K; e += 256) {
    int c = e % 128, k = e / 128;
    int col = n0base + c;
    lBT[c * K + k] = (col < N) ? Bw[(size_t)k * N + col] : (_Float16)0;
  }
  __syncthreads();
  int wave = threadIdx.x >> 5, lane = threadIdx.x & 31;
  int n0 = n0base + wave * 16;
  if (n0 >= N) return;
  int mrow = lane & 15, grp = lane >> 4;
  int lcol = wave * 16 + (lane & 15);
  v8f c = {};
#pragma unroll
  for (int kb = 0; kb < K; kb += 32) {
    // A 16x32 f16 frag: two contiguous 8-half runs per lane
    v8h alo = *(const v8h*)&lA[mrow * K + kb + grp * 8];
    v8h ahi = *(const v8h*)&lA[mrow * K + kb + 16 + grp * 8];
    v16h a = __builtin_shufflevector(alo, ahi, 0, 1, 2, 3, 4, 5, 6, 7,
                                     8, 9, 10, 11, 12, 13, 14, 15);
    // B 32x16 f16 frag: two contiguous 8-half runs per lane in transposed tile
    v8h blo = *(const v8h*)&lBT[lcol * K + kb + grp * 8];
    v8h bhi = *(const v8h*)&lBT[lcol * K + kb + 16 + grp * 8];
    v16h b = __builtin_shufflevector(blo, bhi, 0, 1, 2, 3, 4, 5, 6, 7,
                                     8, 9, 10, 11, 12, 13, 14, 15);
    c = __builtin_amdgcn_wmma_f32_16x16x32_f16(false, a, false, b, (short)0, c,
                                               false, false);
  }
  // unconditional stores: C buffer rows are padded to a multiple of 16
  int ncol = n0 + (lane & 15);
#pragma unroll
  for (int r = 0; r < 8; ++r)
    Cc[(size_t)(m0 + r + 8 * grp) * N + ncol] = c[r];
}

// ---------------- windowed attention, wave per output position ----------------
template <int KH, int KW>
__global__ void attn_kernel(const float* __restrict__ proj, const float* __restrict__ emb,
                            float* __restrict__ outp,
                            int Bn, int Hp, int Wp, int Hout, int Wout,
                            int pt, int pl, int Co, int Nc, int kcol, int vcol) {
  constexpr int KT = KH * KW;
  int wave = threadIdx.x >> 5, lane = threadIdx.x & 31;
  int pos = blockIdx.x * 8 + wave;
  int total = Bn * Hout * Wout;
  if (pos >= total) return;
  int w = pos % Wout;
  int h = (pos / Wout) % Hout;
  int b = pos / (Wout * Hout);

  size_t qrow = ((size_t)(b * Hp + h + pt) * Wp + (w + pl));
  float s[KT];
#pragma unroll
  for (int t = 0; t < KT; ++t) s[t] = 0.f;
  for (int o = lane; o < Co; o += 32) {
    float qv = proj[qrow * Nc + o];                    // q occupies cols [0,Co)
#pragma unroll
    for (int t = 0; t < KT; ++t) {
      int i = t / KW, j = t % KW;
      size_t krow = ((size_t)(b * Hp + h + i) * Wp + (w + j));
      s[t] += qv * proj[krow * Nc + kcol + o];
    }
  }
#pragma unroll
  for (int t = 0; t < KT; ++t)
    for (int off = 16; off > 0; off >>= 1)
      s[t] += __shfl_xor(s[t], off, 32);
  float mx = s[0];
#pragma unroll
  for (int t = 1; t < KT; ++t) mx = fmaxf(mx, s[t]);
  float se = 0.f;
#pragma unroll
  for (int t = 0; t < KT; ++t) { s[t] = __expf(s[t] - mx); se += s[t]; }
  float inv = 1.f / se;

  float embv[4][KT];
#pragma unroll
  for (int m = 0; m < 4; ++m)
#pragma unroll
    for (int t = 0; t < KT; ++t)
      embv[m][t] = emb[(m * KH + t / KW) * KW + (t % KW)];

  for (int o = lane; o < Co; o += 32) {
    float acc = 0.f;
#pragma unroll
    for (int t = 0; t < KT; ++t) {
      int i = t / KW, j = t % KW;
      size_t vrow = ((size_t)(b * Hp + h + i) * Wp + (w + j));
      float vm = 0.f;
#pragma unroll
      for (int m = 0; m < 4; ++m)
        vm += embv[m][t] * proj[vrow * Nc + vcol + m * Co + o];
      acc += s[t] * inv * vm;
    }
    outp[(size_t)pos * Co + o] = acc;
  }
}

// ---------------- final BN + ReLU + AvgPool(1,56) ----------------
__global__ void final_pool(const float* __restrict__ s2, const float* __restrict__ stats,
                           const float* __restrict__ g, const float* __restrict__ bb,
                           float* __restrict__ out) {
  int idx = blockIdx.x * 256 + threadIdx.x;
  if (idx >= 2 * 16 * 13) return;                 // [B,C,1,13]
  int j = idx % 13;
  int c = (idx / 13) % 16;
  int b = idx / (13 * 16);
  float acc = 0.f;
  for (int w = 56 * j; w < 56 * j + 56; ++w) {
    float v = s2[((size_t)b * 750 + w) * 16 + c];
    v = (v - stats[c]) * stats[16 + c] * g[c] + bb[c];
    acc += fmaxf(v, 0.f);
  }
  out[idx] = acc * (1.f / 56.f);
}

extern "C" void kernel_launch(void* const* d_in, const int* in_sizes, int n_in,
                              void* d_out, int out_size, void* d_ws, size_t ws_size,
                              hipStream_t stream) {
  (void)in_sizes; (void)n_in; (void)out_size; (void)ws_size;
  const float* x   = (const float*)d_in[0];
  const float* q0  = (const float*)d_in[1];
  const float* k0  = (const float*)d_in[2];
  const float* v0  = (const float*)d_in[3];
  const float* ea0 = (const float*)d_in[4];
  const float* eb0 = (const float*)d_in[5];
  const float* em0 = (const float*)d_in[6];
  const float* g0  = (const float*)d_in[7];
  const float* b0  = (const float*)d_in[8];
  const float* q1  = (const float*)d_in[9];
  const float* k1  = (const float*)d_in[10];
  const float* v1  = (const float*)d_in[11];
  const float* ea1 = (const float*)d_in[12];
  const float* eb1 = (const float*)d_in[13];
  const float* em1 = (const float*)d_in[14];
  const float* g1  = (const float*)d_in[15];
  const float* b1  = (const float*)d_in[16];
  const float* q2  = (const float*)d_in[17];
  const float* k2  = (const float*)d_in[18];
  const float* v2  = (const float*)d_in[19];
  const float* ea2 = (const float*)d_in[20];
  const float* eb2 = (const float*)d_in[21];
  const float* em2 = (const float*)d_in[22];
  const float* g2  = (const float*)d_in[23];
  const float* b2  = (const float*)d_in[24];

  float* ws = (float*)d_ws;
  _Float16* W1  = (_Float16*)(ws + O_W1);
  _Float16* W2  = (_Float16*)(ws + O_W2);
  _Float16* h1p = (_Float16*)(ws + O_H1P);
  _Float16* h2p = (_Float16*)(ws + O_H2P);

  prep_small<<<1, 64, 0, stream>>>(q0, k0, v0, ea0, eb0, em0,
                                   ea1, eb1, em1, ea2, eb2, em2, ws);
  pack_w<<<240, 256, 0, stream>>>(q1, k1, v1, q2, k2, v2, W1, W2);

  // stage 0: [2,1,8,3000] -> s0 NHWC [2,8,3000,64]
  stage0_attn<<<6000, 256, 0, stream>>>(x, ws, ws + O_S0);
  bn_stats<<<64, 256, 0, stream>>>(ws + O_S0, 48000, 64, ws + O_ST0);
  // BN+ReLU+subsample+pad -> h1p f16 [2,6,1504,64]
  build_pad<<<4512, 256, 0, stream>>>(ws + O_S0, ws + O_ST0, g0, b0, h1p,
                                      2, 8, 3000, 64, 6, 1504, 1, 2, 4, 1500);
  // stage 1 projections: [18048 x 64] x [64 x 768] -> proj1
  gemm_f16<64><<<dim3(1128, 6), 256, 0, stream>>>(h1p, W1, ws + O_PJ1, 18048, 768);
  // stage 1 attention -> s1 NHWC [2,4,1500,128]
  attn_kernel<2, 5><<<1500, 256, 0, stream>>>(ws + O_PJ1, ws + O_EMB1, ws + O_S1,
                                              2, 6, 1504, 4, 1500, 1, 2, 128, 768, 128, 256);
  bn_stats<<<128, 256, 0, stream>>>(ws + O_S1, 12000, 128, ws + O_ST1);
  // BN+ReLU+subsample+pad -> h2p f16 [2,2,754,128]
  build_pad<<<3016, 256, 0, stream>>>(ws + O_S1, ws + O_ST1, g1, b1, h2p,
                                      2, 4, 1500, 128, 2, 754, 0, 2, 2, 750);
  // stage 2 projections: [3016 x 128] x [128 x 96] -> proj2 (rows padded to 3024)
  gemm_f16<128><<<dim3(189, 1), 256, 0, stream>>>(h2p, W2, ws + O_PJ2, 3016, 96);
  // stage 2 attention -> s2 NHWC [2,1,750,16]
  attn_kernel<2, 5><<<188, 256, 0, stream>>>(ws + O_PJ2, ws + O_EMB2, ws + O_S2,
                                             2, 2, 754, 1, 750, 0, 2, 16, 96, 16, 32);
  bn_stats<<<16, 256, 0, stream>>>(ws + O_S2, 1500, 16, ws + O_ST2);
  final_pool<<<2, 256, 0, stream>>>(ws + O_S2, ws + O_ST2, g2, b2, (float*)d_out);
}
